// PointTransformerBlock_56538949485051
// MI455X (gfx1250) — matrix operand order repaired and checked by reading, not compiled, not run on hardware
//
#include <hip/hip_runtime.h>

// Point Transformer block, fully fused, bf16 WMMA (gfx1250 / CDNA5).
// B=8, C_IN=256, C_MID=64, C_CO=3, N=4096, K=16.
// - weights converted once to bf16 in LDS (A operands = 2x ds_load_b128)
// - knn chunks double-buffered in LDS as packed bf16 (B operands = 2x ds_load_b128)
// - all bf16 packing via v_cvt_pk_bf16_f32-friendly pair stores (ds_store_b32)

#define BATCH 8
#define CIN   256
#define CMID  64
#define CCO   3
#define NPTS  4096
#define KNN   16
#define TP    16            // points per workgroup
#define NCOL  (TP * KNN)    // 256 neighbor columns per workgroup

typedef __bf16 bf16_t;
typedef __attribute__((ext_vector_type(16))) __bf16 v16bf;
typedef __attribute__((ext_vector_type(8)))  float  v8f;

// ---- LDS layout (bytes) --------------------------------------------------
#define O_WL1   0         // W_lin1   64x256 bf16 (32 KB)
#define O_WL12  32768     // W_lin1_2 64x256 bf16 (32 KB)
#define O_WL2   65536     // W_lin2  256x64  bf16 (32 KB)
#define O_WPHI  98304     // 64x64 bf16 (8 KB each) ...
#define O_WPSI  106496
#define O_WALP  114688
#define O_WG1   122880
#define O_WG2   131072
#define O_WPE2  139264
#define O_R0    147456    // Xcb double buffer (2 x 16 KB) / Hb alias (32 KB)
#define O_KP    180224    // knn_proj col-major [col][64] (32 KB)
#define O_DL    212992    // delta    col-major (32 KB)
#define O_G     245760    // relu(pe1) / g_in / gamma / rho (32 KB)
#define O_XB    278528    // x   [16 pts][64] (2 KB)
#define O_PHIB  280576    // phi [16 pts][64] (2 KB)
#define O_AGGB  282624    // agg [16 pts][64] (2 KB)
#define SMEM_BYTES 284672 // 278 KB of the WGP's 320 KB LDS

// Pack two floats into one dword of two bf16 (lo, hi) -> v_cvt_pk_bf16_f32.
__device__ __forceinline__ unsigned pk2(float lo, float hi) {
  union { bf16_t h; unsigned short s; } a, b;
  a.h = (bf16_t)lo; b.h = (bf16_t)hi;
  return ((unsigned)b.s << 16) | (unsigned)a.s;
}

// Store 8 accumulator rows (consecutive channels) as 4 packed dwords.
__device__ __forceinline__ void store8_pk(bf16_t* p, const v8f& acc) {
  unsigned* d = (unsigned*)p;   // p is 4B aligned (even bf16 offset)
#pragma unroll
  for (int j = 0; j < 4; ++j) d[j] = pk2(acc[2 * j], acc[2 * j + 1]);
}

// A operand: 16x32 bf16 tile from a row-major bf16 LDS weight [*, ld].
// k order used here (contiguous, k = 16h + s): lanes 0-15 hold rows M0-15 with K 0..15;
//             lanes 16-31 hold K 16..31.
__device__ __forceinline__ v16bf load_a_lds(const bf16_t* W, int ld,
                                            int m0, int k0, int lane) {
  const int row = m0 + (lane & 15);
  const int h   = lane >> 4;
  const bf16_t* p = W + row * ld + k0 + 16 * h;
  v16bf a;
#pragma unroll
  for (int j = 0; j < 8; ++j) a[j]     = p[j];
#pragma unroll
  for (int j = 0; j < 8; ++j) a[8 + j] = p[8 + j];
  return a;
}

// B operand: 32x16 bf16 tile from a column-major LDS buffer X[col][stride].
// Lanes 0-15 -> K 0..15 of column n; lanes 16-31 -> K 16..31.
__device__ __forceinline__ v16bf load_b_cm(const bf16_t* X, int col, int c0,
                                           int stride, int lane) {
  const int h = lane >> 4;
  const bf16_t* p = X + col * stride + c0 + 16 * h;
  v16bf bv;
#pragma unroll
  for (int j = 0; j < 16; ++j) bv[j] = p[j];
  return bv;
}

__device__ __forceinline__ v8f wmma_bf16(v16bf a, v16bf b, v8f c) {
  return __builtin_amdgcn_wmma_f32_16x16x32_bf16(false, a, false, b,
                                                 (short)0, c, false, false);
}

// One 16x16 output tile of [64 x cols] = W[64x64] @ X[64 x cols],
// contraction split into two K=32 WMMA steps. W and X are bf16 in LDS.
__device__ __forceinline__ v8f gemm64_tile(const bf16_t* W, const bf16_t* src,
                                           int mt, int col, int lane) {
  v8f acc = {};
#pragma unroll
  for (int s = 0; s < 2; ++s) {
    v16bf a  = load_a_lds(W, CMID, 16 * mt, 32 * s, lane);
    v16bf bv = load_b_cm(src, col, 32 * s, CMID, lane);
    acc = wmma_bf16(a, bv, acc);
  }
  return acc;
}

__global__ __launch_bounds__(256, 1)
void pt_block_fused(const float* __restrict__ pooled,
                    const float* __restrict__ knn,
                    const float* __restrict__ lcoord,
                    const float* __restrict__ W_lin1,
                    const float* __restrict__ W_lin12,
                    const float* __restrict__ W_lin2,
                    const float* __restrict__ W_phi,
                    const float* __restrict__ W_psi,
                    const float* __restrict__ W_alpha,
                    const float* __restrict__ W_g1,
                    const float* __restrict__ W_g2,
                    const float* __restrict__ W_pe1,
                    const float* __restrict__ W_pe2,
                    float* __restrict__ out) {
  extern __shared__ char smem[];
  bf16_t* WL1  = (bf16_t*)(smem + O_WL1);
  bf16_t* WL12 = (bf16_t*)(smem + O_WL12);
  bf16_t* WL2  = (bf16_t*)(smem + O_WL2);
  bf16_t* WPHI = (bf16_t*)(smem + O_WPHI);
  bf16_t* WPSI = (bf16_t*)(smem + O_WPSI);
  bf16_t* WALP = (bf16_t*)(smem + O_WALP);
  bf16_t* WG1  = (bf16_t*)(smem + O_WG1);
  bf16_t* WG2  = (bf16_t*)(smem + O_WG2);
  bf16_t* WPE2 = (bf16_t*)(smem + O_WPE2);
  bf16_t* Xb0  = (bf16_t*)(smem + O_R0);          // chunk buf 0 [col][32]
  bf16_t* Xb1  = (bf16_t*)(smem + O_R0 + 16384);  // chunk buf 1 [col][32]
  bf16_t* Hb   = (bf16_t*)(smem + O_R0);          // alias: relu(g1) / val
  bf16_t* KP   = (bf16_t*)(smem + O_KP);
  bf16_t* DL   = (bf16_t*)(smem + O_DL);
  bf16_t* G    = (bf16_t*)(smem + O_G);
  bf16_t* xb   = (bf16_t*)(smem + O_XB);
  bf16_t* phib = (bf16_t*)(smem + O_PHIB);
  bf16_t* aggb = (bf16_t*)(smem + O_AGGB);

  const int tid  = threadIdx.x;
  const int lane = tid & 31;
  const int wave = tid >> 5;       // 0..7
  const int ln   = lane & 15;      // N index inside tile
  const int lh   = lane >> 4;      // lane half
  const int bpB  = NPTS / TP;
  const int b    = blockIdx.x / bpB;
  const int p0   = (blockIdx.x % bpB) * TP;

  // ---- Stage W: convert all weights to bf16 in LDS (once per block) ----
  for (int i = tid; i < CMID * CIN; i += 256) {
    WL1[i]  = (bf16_t)W_lin1[i];
    WL12[i] = (bf16_t)W_lin12[i];
    WL2[i]  = (bf16_t)W_lin2[i];     // 256x64 = same element count
  }
  for (int i = tid; i < CMID * CMID; i += 256) {
    WPHI[i] = (bf16_t)W_phi[i];
    WPSI[i] = (bf16_t)W_psi[i];
    WALP[i] = (bf16_t)W_alpha[i];
    WG1[i]  = (bf16_t)W_g1[i];
    WG2[i]  = (bf16_t)W_g2[i];
    WPE2[i] = (bf16_t)W_pe2[i];
  }
  __syncthreads();

  // ---- Stage A: x = W_lin1 @ pooled  (64 x 16 points) ------------------
  // staging assignment: channel pair -> packed dword stores
  const int cpA  = tid >> 4;        // channel pair 0..15 (ch 2cpA, 2cpA+1)
  const int colA = tid & 15;        // point column
  v8f accA = {};
  for (int s = 0; s < CIN / 32; ++s) {
    const int c0 = 32 * s;
    {
      const float* pA =
          pooled + ((size_t)(b * CIN + c0 + 2 * cpA)) * NPTS + p0 + colA;
      float lo = pA[0];
      float hi = pA[NPTS];
      ((unsigned*)Xb0)[colA * 16 + cpA] = pk2(lo, hi);
    }
    __syncthreads();
    if (wave < 4) {
      v16bf a  = load_a_lds(WL1, CIN, 16 * wave, c0, lane);
      v16bf bv = load_b_cm(Xb0, ln, 0, 32, lane);
      accA = wmma_bf16(a, bv, accA);
    }
    __syncthreads();
  }
  if (wave < 4)
    store8_pk(xb + ln * CMID + 16 * wave + 8 * lh, accA);
  __syncthreads();

  // ---- Stage B: phi = W_phi @ x ----------------------------------------
  if (wave < 4) {
    v8f acc = gemm64_tile(WPHI, xb, wave, ln, lane);
    store8_pk(phib + ln * CMID + 16 * wave + 8 * lh, acc);
  }
  __syncthreads();

  // ---- Stage C: delta = W_pe2 @ relu(W_pe1 @ local_coords) -------------
  // C1: pe1 with K=3 zero-padded into a single K=32 WMMA step.
#pragma unroll
  for (int t = 0; t < 2; ++t) {
    const int nt  = 2 * wave + t;
    const int col = nt * 16 + ln;
#pragma unroll
    for (int mt = 0; mt < 4; ++mt) {
      v16bf a;
#pragma unroll
      for (int j = 0; j < 8; ++j) {
        int k = 8 * lh + j;
        a[j] = (k < CCO) ? (bf16_t)W_pe1[(16 * mt + ln) * CCO + k] : (bf16_t)0.f;
      }
#pragma unroll
      for (int j = 0; j < 8; ++j) a[8 + j] = (bf16_t)0.f;
      v16bf bv;
#pragma unroll
      for (int j = 0; j < 16; ++j) {
        int c = 16 * lh + j;
        bv[j] = (c < CCO)
            ? (bf16_t)lcoord[(((size_t)(b * CCO + c)) * NPTS + p0 + nt) * KNN + ln]
            : (bf16_t)0.f;
      }
      v8f zc  = {};
      v8f acc = wmma_bf16(a, bv, zc);
      v8f rl;
#pragma unroll
      for (int r = 0; r < 8; ++r) rl[r] = fmaxf(acc[r], 0.f);
      store8_pk(G + col * CMID + 16 * mt + 8 * lh, rl);
    }
  }
  __syncthreads();
  // C2: pe2
#pragma unroll
  for (int t = 0; t < 2; ++t) {
    const int col = (2 * wave + t) * 16 + ln;
#pragma unroll
    for (int mt = 0; mt < 4; ++mt) {
      v8f acc = gemm64_tile(WPE2, G, mt, col, lane);
      store8_pk(DL + col * CMID + 16 * mt + 8 * lh, acc);
    }
  }
  __syncthreads();

  // ---- Stage D: knn_proj = W_lin1_2 @ knn_features (K=256 streamed) ----
  // Double-buffered chunk staging: global loads for chunk s+1 are issued
  // before the WMMAs of chunk s; one barrier per k-step.
  v8f accD[4][2] = {};
  const int cp   = tid >> 4;          // channel pair 0..15
  const int colb = (tid & 15) * 16;   // column base (16 cols per thread)
  const size_t chstride = (size_t)32 * NPTS * KNN;       // 32 channels
  const float* base0 =
      knn + (((size_t)(b * CIN + 2 * cp)) * NPTS + p0) * KNN + colb;
  const float* base1 = base0 + (size_t)NPTS * KNN;       // next channel

  float r0[16], r1[16];
  bf16_t* Xb[2] = {Xb0, Xb1};

  auto ldchunk = [&](int s) {
    const float4* s0 = (const float4*)(base0 + s * chstride);
    const float4* s1 = (const float4*)(base1 + s * chstride);
#pragma unroll
    for (int i = 0; i < 4; ++i) {
      float4 v = s0[i];
      r0[4 * i] = v.x; r0[4 * i + 1] = v.y; r0[4 * i + 2] = v.z; r0[4 * i + 3] = v.w;
      float4 w = s1[i];
      r1[4 * i] = w.x; r1[4 * i + 1] = w.y; r1[4 * i + 2] = w.z; r1[4 * i + 3] = w.w;
    }
  };
  auto stchunk = [&](int buf) {
    unsigned* d = (unsigned*)Xb[buf];
#pragma unroll
    for (int i = 0; i < 16; ++i)
      d[(colb + i) * 16 + cp] = pk2(r0[i], r1[i]);
  };

  ldchunk(0);
  stchunk(0);
  __syncthreads();
  for (int s = 0; s < CIN / 32; ++s) {
    if (s + 1 < CIN / 32) ldchunk(s + 1);                  // overlap with WMMA
    if (s + 2 < CIN / 32)
      __builtin_prefetch(base0 + (s + 2) * chstride, 0, 1);
    const bf16_t* Xcur = Xb[s & 1];
#pragma unroll
    for (int mt = 0; mt < 4; ++mt) {
      v16bf a = load_a_lds(WL12, CIN, 16 * mt, 32 * s, lane);
#pragma unroll
      for (int t = 0; t < 2; ++t) {
        const int col = (2 * wave + t) * 16 + ln;
        v16bf bv = load_b_cm(Xcur, col, 0, 32, lane);
        accD[mt][t] = wmma_bf16(a, bv, accD[mt][t]);
      }
    }
    if (s + 1 < CIN / 32) stchunk((s + 1) & 1);
    __syncthreads();
  }
#pragma unroll
  for (int t = 0; t < 2; ++t) {
    const int col = (2 * wave + t) * 16 + ln;
#pragma unroll
    for (int mt = 0; mt < 4; ++mt)
      store8_pk(KP + col * CMID + 16 * mt + 8 * lh, accD[mt][t]);
  }
  __syncthreads();

  // ---- Stage E: psi = W_psi @ KP ; g_in = phi - psi + delta ------------
#pragma unroll
  for (int t = 0; t < 2; ++t) {
    const int nt  = 2 * wave + t;
    const int col = nt * 16 + ln;
#pragma unroll
    for (int mt = 0; mt < 4; ++mt) {
      v8f acc = gemm64_tile(WPSI, KP, mt, col, lane);
      v8f gv;
#pragma unroll
      for (int r = 0; r < 8; ++r) {
        const int m = 16 * mt + r + 8 * lh;
        gv[r] = (float)phib[nt * CMID + m] - acc[r] + (float)DL[col * CMID + m];
      }
      store8_pk(G + col * CMID + 16 * mt + 8 * lh, gv);
    }
  }
  __syncthreads();

  // ---- Stage F: relu(W_g1 @ g_in) -> Hb --------------------------------
#pragma unroll
  for (int t = 0; t < 2; ++t) {
    const int col = (2 * wave + t) * 16 + ln;
#pragma unroll
    for (int mt = 0; mt < 4; ++mt) {
      v8f acc = gemm64_tile(WG1, G, mt, col, lane);
      v8f rl;
#pragma unroll
      for (int r = 0; r < 8; ++r) rl[r] = fmaxf(acc[r], 0.f);
      store8_pk(Hb + col * CMID + 16 * mt + 8 * lh, rl);
    }
  }
  __syncthreads();

  // ---- Stage G: gamma = W_g2 @ Hb -> G ---------------------------------
#pragma unroll
  for (int t = 0; t < 2; ++t) {
    const int col = (2 * wave + t) * 16 + ln;
#pragma unroll
    for (int mt = 0; mt < 4; ++mt) {
      v8f acc = gemm64_tile(WG2, Hb, mt, col, lane);
      store8_pk(G + col * CMID + 16 * mt + 8 * lh, acc);
    }
  }
  __syncthreads();

  // ---- Stage H: val = W_alpha @ KP + delta -> Hb -----------------------
#pragma unroll
  for (int t = 0; t < 2; ++t) {
    const int col = (2 * wave + t) * 16 + ln;
#pragma unroll
    for (int mt = 0; mt < 4; ++mt) {
      v8f acc = gemm64_tile(WALP, KP, mt, col, lane);
      v8f vv;
#pragma unroll
      for (int r = 0; r < 8; ++r) {
        const int m = 16 * mt + r + 8 * lh;
        vv[r] = acc[r] + (float)DL[col * CMID + m];
      }
      store8_pk(Hb + col * CMID + 16 * mt + 8 * lh, vv);
    }
  }
  __syncthreads();

  // ---- Stage I: softmax over 16 neighbors + weighted sum ---------------
#pragma unroll
  for (int i = 0; i < 4; ++i) {
    const int g = tid + 256 * i;          // 1024 (channel, point) groups
    const int m = g & 63;
    const int p = g >> 6;
    float e[16], mx = -3.0e38f;
#pragma unroll
    for (int kk = 0; kk < 16; ++kk) {
      e[kk] = (float)G[(p * 16 + kk) * CMID + m];
      mx = fmaxf(mx, e[kk]);
    }
    float se = 0.f, av = 0.f;
#pragma unroll
    for (int kk = 0; kk < 16; ++kk) {
      float w = __expf(e[kk] - mx);
      se += w;
      av += w * (float)Hb[(p * 16 + kk) * CMID + m];
    }
    aggb[p * CMID + m] = (bf16_t)(av / se);
  }
  __syncthreads();

  // ---- Stage J: out = W_lin2 (256x64) @ agg (64x16) --------------------
#pragma unroll
  for (int jj = 0; jj < 2; ++jj) {
    const int mt = 2 * wave + jj;         // 16 output M-tiles over 8 waves
    v8f acc = {};
#pragma unroll
    for (int s = 0; s < 2; ++s) {
      v16bf a  = load_a_lds(WL2, CMID, 16 * mt, 32 * s, lane);
      v16bf bv = load_b_cm(aggb, ln, 32 * s, CMID, lane);
      acc = wmma_bf16(a, bv, acc);
    }
#pragma unroll
    for (int r = 0; r < 8; ++r) {
      const int mo = 16 * mt + r + 8 * lh;
      out[((size_t)b * CIN + mo) * NPTS + p0 + ln] = acc[r];
    }
  }
}

extern "C" void kernel_launch(void* const* d_in, const int* in_sizes, int n_in,
                              void* d_out, int out_size, void* d_ws, size_t ws_size,
                              hipStream_t stream) {
  (void)in_sizes; (void)n_in; (void)out_size; (void)d_ws; (void)ws_size;
  const float* pooled  = (const float*)d_in[0];
  // d_in[1] = coords: unused by the reference computation.
  const float* knn     = (const float*)d_in[2];
  const float* lcoord  = (const float*)d_in[3];
  const float* W_lin1  = (const float*)d_in[4];
  const float* W_lin12 = (const float*)d_in[5];
  const float* W_lin2  = (const float*)d_in[6];
  const float* W_phi   = (const float*)d_in[7];
  const float* W_psi   = (const float*)d_in[8];
  const float* W_alpha = (const float*)d_in[9];
  const float* W_g1    = (const float*)d_in[10];
  const float* W_g2    = (const float*)d_in[11];
  const float* W_pe1   = (const float*)d_in[12];
  const float* W_pe2   = (const float*)d_in[13];
  float* out = (float*)d_out;

  hipFuncSetAttribute((const void*)pt_block_fused,
                      hipFuncAttributeMaxDynamicSharedMemorySize, SMEM_BYTES);

  dim3 grid(BATCH * (NPTS / TP));   // 8 batches x 256 point-tiles = 2048 WGs
  pt_block_fused<<<grid, 256, SMEM_BYTES, stream>>>(
      pooled, knn, lcoord, W_lin1, W_lin12, W_lin2, W_phi, W_psi, W_alpha,
      W_g1, W_g2, W_pe1, W_pe2, out);
}
